// Transformer_7722351199135
// MI455X (gfx1250) — matrix operand order repaired and checked by reading, not compiled, not run on hardware
//
#include <hip/hip_runtime.h>
#include <hip/hip_bf16.h>

typedef __attribute__((ext_vector_type(16))) _Float16 v16h;
typedef __attribute__((ext_vector_type(8)))  _Float16 v8h;
typedef __attribute__((ext_vector_type(8)))  float    v8f;
typedef __attribute__((ext_vector_type(4)))  unsigned int u32x4;
typedef __attribute__((ext_vector_type(8)))  int          i32x8;
typedef __attribute__((ext_vector_type(4)))  int          i32x4;

#define NB     8
#define TSEQ   512
#define DMODEL 512
#define NH     8
#define DHEAD  64
#define DFF    2048
#define NLE    6
#define NLD    6
#define LNEPS  1e-5f

// Tensor Data Mover availability (this toolchain: clang-23, 6-arg builtin).
#if defined(__has_builtin)
#if __has_builtin(__builtin_amdgcn_tensor_load_to_lds) && \
    __has_builtin(__builtin_amdgcn_s_wait_tensorcnt)
#define USE_TDM 1
#endif
#endif
#ifndef USE_TDM
#define USE_TDM 0
#endif

#if USE_TDM
// 2D tile load global->LDS via TDM (CDNA5 ISA D# layout).
// Copies tile_rows rows of tile_k f16 elements; row stride = ld elements.
// LDS gets 16B padding after every 128B row -> LDS row stride 72 f16.
__device__ __forceinline__ void tdm_load_2d(
    unsigned long long gaddr, unsigned int lds_off,
    unsigned int tile_k, unsigned int tile_rows,
    unsigned long long ld, unsigned int tensor_k, unsigned int tensor_rows)
{
    u32x4 g0;
    g0[0] = 1u;                                       // count=1, user mode
    g0[1] = lds_off;                                  // lds_addr (bytes)
    g0[2] = (unsigned int)(gaddr & 0xFFFFFFFFull);    // global_addr[31:0]
    g0[3] = (unsigned int)((gaddr >> 32) & 0x01FFFFFFull) | 0x80000000u; // +type=2
    i32x8 g1;
    // data_size=1 (2B), pad_enable, pad_interval=4 (32 DW = 128B),
    // pad_amount=3 (4 DW = 16B), wg_mask=0
    g1[0] = (int)((1u << 16) | (1u << 20) | (4u << 22) | (3u << 25));
    g1[1] = (int)((tensor_k & 0xFFFFu) << 16);                     // dim0 lo16
    g1[2] = (int)((tensor_k >> 16) | ((tensor_rows & 0xFFFFu) << 16));
    g1[3] = (int)((tensor_rows >> 16) | (tile_k << 16));           // tile_dim0
    g1[4] = (int)(tile_rows & 0xFFFFu);                            // tile_dim1
    g1[5] = (int)(ld & 0xFFFFFFFFull);                             // stride lo32
    g1[6] = (int)((ld >> 32) & 0xFFFFull);                         // stride hi16
    g1[7] = 0;
    i32x4 z4; z4[0] = 0; z4[1] = 0; z4[2] = 0; z4[3] = 0;
    i32x8 z8;
#pragma unroll
    for (int j = 0; j < 8; j++) z8[j] = 0;
    __builtin_amdgcn_tensor_load_to_lds(g0, g1, z4, z4, z8, 0);
}
#endif

// ---------------------------------------------------------------------------
// Batched WMMA GEMM:  C = scale * (A x op(B)) + bias, optional ReLU.
//   A: (M,K) f16 row-major, lda
//   transB=1 -> B[k][n] = Bm[n*ldb + k]  (N-major; weights are pre-transposed)
//   transB=0 -> B[k][n] = Bm[k*ldb + n]  (attn.V only)
// Block: 256 threads = 8 wave32 waves; 128x128 tile, 64-wide K stages.
// Each wave: 64(M) x 32(N); 16 v_wmma_f32_16x16x32_f16 per stage.
// A tile (and B tile when transB) staged by one tensor_load_to_lds each.
// ---------------------------------------------------------------------------
__global__ __launch_bounds__(256) void wmma_gemm_f16(
    const _Float16* __restrict__ A, const _Float16* __restrict__ Bm,
    float* __restrict__ C32, _Float16* __restrict__ C16,
    const float* __restrict__ bias,
    int M, int N, int K,
    long long lda, long long ldb, long long ldc,
    long long sAb, long long sAh, long long sBb, long long sBh,
    long long sCb, long long sCh,
    int hdiv, int transB, int relu, float scale)
{
    __shared__ _Float16 As[128 * 72];   // [m][k0..63], row stride 72 (144B)
    __shared__ _Float16 Bs[128 * 72];   // [n][k0..63], row stride 72

    const int tid  = threadIdx.x;
    const int lane = tid & 31;
    const int wave = tid >> 5;
    const int wM   = wave >> 2;   // 0..1
    const int wN   = wave & 3;    // 0..3
    const int ln15 = lane & 15;
    const int hi   = lane >> 4;

    const int m0 = blockIdx.y * 128;
    const int n0 = blockIdx.x * 128;
    const int z  = blockIdx.z;
    const int zb = z / hdiv;
    const int zh = z - zb * hdiv;

    const _Float16* Ab = A  + zb * sAb + zh * sAh;
    const _Float16* Bb = Bm + zb * sBb + zh * sBh;

    v8f acc[4][2];
#pragma unroll
    for (int i = 0; i < 4; i++)
#pragma unroll
        for (int j = 0; j < 2; j++)
#pragma unroll
            for (int r = 0; r < 8; r++) acc[i][j][r] = 0.0f;

    for (int kk = 0; kk < K; kk += 64) {
#if USE_TDM
        if (tid == 0) {
            tdm_load_2d((unsigned long long)(size_t)(Ab + (long long)m0 * lda + kk),
                        (unsigned int)(size_t)As, 64u, 128u,
                        (unsigned long long)lda,
                        (unsigned int)(K - kk), (unsigned int)(M - m0));
            if (transB)
                tdm_load_2d((unsigned long long)(size_t)(Bb + (long long)n0 * ldb + kk),
                            (unsigned int)(size_t)Bs, 64u, 128u,
                            (unsigned long long)ldb,
                            (unsigned int)(K - kk), (unsigned int)(N - n0));
            __builtin_amdgcn_s_wait_tensorcnt(0);
        }
#else
        // ---- manual A tile: 128 rows x 64 k
#pragma unroll
        for (int c = tid; c < 1024; c += 256) {
            int r  = c >> 3;
            int kc = (c & 7) * 8;
            int gm = m0 + r;
            v8h v;
            if (gm < M) {
                v = *(const v8h*)(Ab + (long long)gm * lda + (kk + kc));
            } else {
#pragma unroll
                for (int j = 0; j < 8; j++) v[j] = (_Float16)0.0f;
            }
            *(v8h*)(&As[r * 72 + kc]) = v;
        }
        if (transB) {
#pragma unroll
            for (int c = tid; c < 1024; c += 256) {
                int r  = c >> 3;
                int kc = (c & 7) * 8;
                int gn = n0 + r;
                v8h v;
                if (gn < N) {
                    v = *(const v8h*)(Bb + (long long)gn * ldb + (kk + kc));
                } else {
#pragma unroll
                    for (int j = 0; j < 8; j++) v[j] = (_Float16)0.0f;
                }
                *(v8h*)(&Bs[r * 72 + kc]) = v;
            }
        }
#endif
        if (!transB) {
            // attn.V path: global is k-major, scatter to n-major LDS
#pragma unroll
            for (int c = tid; c < 1024; c += 256) {
                int kr = c >> 4;           // 0..63
                int nc = (c & 15) * 8;     // 0..120
                int gn = n0 + nc;
                _Float16 tmp[8];
                if (gn + 8 <= N) {
                    v8h v = *(const v8h*)(Bb + (long long)(kk + kr) * ldb + gn);
#pragma unroll
                    for (int j = 0; j < 8; j++) tmp[j] = v[j];
                } else {
#pragma unroll
                    for (int j = 0; j < 8; j++)
                        tmp[j] = (gn + j < N)
                                   ? Bb[(long long)(kk + kr) * ldb + gn + j]
                                   : (_Float16)0.0f;
                }
#pragma unroll
                for (int j = 0; j < 8; j++) Bs[(nc + j) * 72 + kr] = tmp[j];
            }
        }
        __syncthreads();

        // ---- 2 x (4x2) WMMA over the 64-wide K stage
#pragma unroll
        for (int kc = 0; kc < 64; kc += 32) {
            v16h afrag[4];
#pragma unroll
            for (int mt = 0; mt < 4; mt++) {
                int row = wM * 64 + mt * 16 + ln15;
                union { v16h v; v8h h[2]; } u;
                u.h[0] = *(const v8h*)(&As[row * 72 + kc + hi * 8]);
                u.h[1] = *(const v8h*)(&As[row * 72 + kc + 16 + hi * 8]);
                afrag[mt] = u.v;
            }
            v16h bfrag[2];
#pragma unroll
            for (int nt = 0; nt < 2; nt++) {
                int col = wN * 32 + nt * 16 + ln15;
                union { v16h v; v8h h[2]; } u;
                u.h[0] = *(const v8h*)(&Bs[col * 72 + kc + hi * 8]);
                u.h[1] = *(const v8h*)(&Bs[col * 72 + kc + 16 + hi * 8]);
                bfrag[nt] = u.v;
            }
#pragma unroll
            for (int mt = 0; mt < 4; mt++)
#pragma unroll
                for (int nt = 0; nt < 2; nt++)
                    acc[mt][nt] = __builtin_amdgcn_wmma_f32_16x16x32_f16(
                        false, afrag[mt], false, bfrag[nt],
                        (short)0, acc[mt][nt], false, false);
        }
        __syncthreads();
    }

    float*    C32b = C32 ? (C32 + zb * sCb + zh * sCh) : (float*)0;
    _Float16* C16b = C16 ? (C16 + zb * sCb + zh * sCh) : (_Float16*)0;
#pragma unroll
    for (int mt = 0; mt < 4; mt++) {
#pragma unroll
        for (int nt = 0; nt < 2; nt++) {
            int gn = n0 + wN * 32 + nt * 16 + ln15;
            if (gn >= N) continue;
            float bv = bias ? bias[gn] : 0.0f;
#pragma unroll
            for (int r = 0; r < 8; r++) {
                int gm = m0 + wM * 64 + mt * 16 + hi * 8 + r;
                if (gm >= M) continue;
                float v = acc[mt][nt][r] * scale + bv;
                if (relu) v = v > 0.0f ? v : 0.0f;
                long long ci = (long long)gm * ldc + gn;
                if (C32b) C32b[ci] = v;
                if (C16b) C16b[ci] = (_Float16)v;
            }
        }
    }
}

// ---------------------------------------------------------------------------
// Column softmax over the QUERY axis (faithful to softmax(scores, axis=2)).
// ---------------------------------------------------------------------------
__global__ __launch_bounds__(256) void softmax_q(
    const float* __restrict__ S, _Float16* __restrict__ Aout,
    int Tq, int Ts, int causal)
{
    int z = blockIdx.x;
    int s = blockIdx.y * blockDim.x + threadIdx.x;
    const float* col = S + (size_t)z * Tq * Ts + s;
    _Float16*    out = Aout + (size_t)z * Tq * Ts + s;

    float m = -3.4e38f;
    for (int q = 0; q < Tq; q++) {
        if (causal && s > q) continue;
        m = fmaxf(m, col[(size_t)q * Ts]);
    }
    float sum = 0.0f;
    for (int q = 0; q < Tq; q++) {
        if (causal && s > q) continue;
        sum += __expf(col[(size_t)q * Ts] - m);
    }
    float inv = 1.0f / sum;
    for (int q = 0; q < Tq; q++) {
        float v = (causal && s > q) ? 0.0f : __expf(col[(size_t)q * Ts] - m) * inv;
        out[(size_t)q * Ts] = (_Float16)v;
    }
}

// ---------------------------------------------------------------------------
// LayerNorm over the full (T, DM) slab per batch (faithful to reference).
// ---------------------------------------------------------------------------
__global__ __launch_bounds__(256) void ln_stats(
    const float* __restrict__ X, const float* __restrict__ R,
    float* __restrict__ stats, int n)
{
    int b = blockIdx.x;
    const float* x = X + (size_t)b * n;
    const float* r = R + (size_t)b * n;
    float s = 0.0f, s2 = 0.0f;
    for (int i = threadIdx.x; i < n; i += 256) {
        float v = x[i] + r[i];
        s += v; s2 += v * v;
    }
    __shared__ float sh0[256], sh1[256];
    sh0[threadIdx.x] = s; sh1[threadIdx.x] = s2;
    __syncthreads();
    for (int o = 128; o > 0; o >>= 1) {
        if (threadIdx.x < o) {
            sh0[threadIdx.x] += sh0[threadIdx.x + o];
            sh1[threadIdx.x] += sh1[threadIdx.x + o];
        }
        __syncthreads();
    }
    if (threadIdx.x == 0) {
        float mu  = sh0[0] / (float)n;
        float var = sh1[0] / (float)n - mu * mu;
        stats[b * 2 + 0] = mu;
        stats[b * 2 + 1] = rsqrtf(var + LNEPS);
    }
}

__global__ __launch_bounds__(256) void ln_apply(
    const float* __restrict__ X, const float* __restrict__ R,
    const float* __restrict__ stats,
    float* __restrict__ O32, _Float16* __restrict__ O16, int n)
{
    int b = blockIdx.y;
    size_t base = (size_t)b * n;
    float mu = stats[b * 2 + 0];
    float rs = stats[b * 2 + 1];
    for (int i = blockIdx.x * blockDim.x + threadIdx.x; i < n;
         i += gridDim.x * blockDim.x) {
        float v = (X[base + i] + R[base + i] - mu) * rs;
        O32[base + i] = v;
        O16[base + i] = (_Float16)v;
    }
}

// ---------------------------------------------------------------------------
// Embedding gather: writes f32 (residual path) and f16 (GEMM path).
// ---------------------------------------------------------------------------
__global__ __launch_bounds__(256) void embed_gather(
    const int* __restrict__ ids, const float* __restrict__ emb,
    float* __restrict__ O32, _Float16* __restrict__ O16, long long total)
{
    long long i = blockIdx.x * (long long)blockDim.x + threadIdx.x;
    if (i >= total) return;
    long long t = i / DMODEL;
    int d = (int)(i - t * DMODEL);
    float v = emb[(long long)ids[t] * DMODEL + d];
    O32[i] = v;
    O16[i] = (_Float16)v;
}

// ---------------------------------------------------------------------------
// Weight repack (L,H,DM,DK) f32 -> (L, H*DK, DM) f16  [N-major / transposed,
// so every weight GEMM uses the TDM-friendly transB path].
// ---------------------------------------------------------------------------
__global__ __launch_bounds__(256) void pack_proj_f16(
    const float* __restrict__ W, _Float16* __restrict__ O, int layers)
{
    long long i = blockIdx.x * (long long)blockDim.x + threadIdx.x;
    long long total = (long long)layers * DMODEL * NH * DHEAD;
    if (i >= total) return;
    const int per = DMODEL * NH * DHEAD;
    int l   = (int)(i / per);
    int rem = (int)(i - (long long)l * per);
    int n = rem / DMODEL;            // n = h*DHEAD + k
    int d = rem - n * DMODEL;
    int h = n / DHEAD, k = n - h * DHEAD;
    O[i] = (_Float16)W[(long long)l * per + (long long)h * DMODEL * DHEAD +
                       (long long)d * DHEAD + k];
}

// (L,K,N) f32 -> (L,N,K) f16 transpose-convert for Wo/W1/W2.
__global__ __launch_bounds__(256) void cvt_t_f32_f16(
    const float* __restrict__ X, _Float16* __restrict__ O,
    int K, int N, int layers)
{
    long long i = blockIdx.x * (long long)blockDim.x + threadIdx.x;
    long long total = (long long)layers * K * N;
    if (i >= total) return;
    long long per = (long long)K * N;
    int l = (int)(i / per);
    long long rem = i - (long long)l * per;
    int n = (int)(rem / K);
    int k = (int)(rem - (long long)n * K);
    O[i] = (_Float16)X[(long long)l * per + (long long)k * N + n];
}

// ---------------------------------------------------------------------------
// Host-side helpers
// ---------------------------------------------------------------------------
static inline void launch_gemm(hipStream_t s,
    const _Float16* A, const _Float16* Bm, float* C32, _Float16* C16,
    const float* bias, int M, int N, int K,
    long long lda, long long ldb, long long ldc,
    long long sAb, long long sAh, long long sBb, long long sBh,
    long long sCb, long long sCh,
    int batches, int hdiv, int transB, int relu, float scale)
{
    dim3 grid((N + 127) / 128, (M + 127) / 128, batches);
    wmma_gemm_f16<<<grid, dim3(256, 1, 1), 0, s>>>(
        A, Bm, C32, C16, bias, M, N, K, lda, ldb, ldc,
        sAb, sAh, sBb, sBh, sCb, sCh, hdiv, transB, relu, scale);
}

extern "C" void kernel_launch(void* const* d_in, const int* in_sizes, int n_in,
                              void* d_out, int out_size, void* d_ws, size_t ws_size,
                              hipStream_t stream)
{
    (void)in_sizes; (void)n_in; (void)out_size; (void)ws_size;

    const int*   src_ids  = (const int*)d_in[0];
    const int*   tgt_ids  = (const int*)d_in[1];
    const float* src_emb  = (const float*)d_in[2];
    const float* tgt_emb  = (const float*)d_in[3];
    const float* enc_Wq   = (const float*)d_in[4];
    const float* enc_Wk   = (const float*)d_in[5];
    const float* enc_Wv   = (const float*)d_in[6];
    const float* enc_Wo   = (const float*)d_in[7];
    const float* enc_W1   = (const float*)d_in[8];
    const float* enc_b1   = (const float*)d_in[9];
    const float* enc_W2   = (const float*)d_in[10];
    const float* enc_b2   = (const float*)d_in[11];
    const float* dec1_Wq  = (const float*)d_in[12];
    const float* dec1_Wk  = (const float*)d_in[13];
    const float* dec1_Wv  = (const float*)d_in[14];
    const float* dec1_Wo  = (const float*)d_in[15];
    const float* dec2_Wq  = (const float*)d_in[16];
    const float* dec2_Wk  = (const float*)d_in[17];
    const float* dec2_Wv  = (const float*)d_in[18];
    const float* dec2_Wo  = (const float*)d_in[19];
    const float* dec_W1   = (const float*)d_in[20];
    const float* dec_b1   = (const float*)d_in[21];
    const float* dec_W2   = (const float*)d_in[22];
    const float* dec_b2   = (const float*)d_in[23];

    const long long NTOK  = (long long)NB * TSEQ;          // 4096
    const long long AELEM = NTOK * DMODEL;
    const long long SELEM = (long long)NB * NH * TSEQ * TSEQ;

    char* wp = (char*)d_ws;
    auto alloc = [&](size_t bytes) -> void* {
        void* r = (void*)wp;
        wp += (bytes + 255) & ~(size_t)255;
        return r;
    };
    float*    x32   = (float*)alloc(AELEM * 4);
    _Float16* x16   = (_Float16*)alloc(AELEM * 2);
    float*    y32   = (float*)alloc(AELEM * 4);
    _Float16* y16   = (_Float16*)alloc(AELEM * 2);
    _Float16* q16   = (_Float16*)alloc(AELEM * 2);
    _Float16* k16   = (_Float16*)alloc(AELEM * 2);
    _Float16* v16   = (_Float16*)alloc(AELEM * 2);
    _Float16* ao16  = (_Float16*)alloc(AELEM * 2);
    float*    sc32  = (float*)alloc(SELEM * 4);
    _Float16* at16  = (_Float16*)alloc(SELEM * 2);
    float*    t32   = (float*)alloc(AELEM * 4);
    _Float16* h16   = (_Float16*)alloc(NTOK * DFF * 2);
    float*    f32b  = (float*)alloc(AELEM * 4);
    float*    stats = (float*)alloc(NB * 2 * 4);

    const size_t PW = (size_t)DMODEL * DMODEL;
    const size_t FW = (size_t)DMODEL * DFF;
    _Float16* eWq  = (_Float16*)alloc(NLE * PW * 2);
    _Float16* eWk  = (_Float16*)alloc(NLE * PW * 2);
    _Float16* eWv  = (_Float16*)alloc(NLE * PW * 2);
    _Float16* eWo  = (_Float16*)alloc(NLE * PW * 2);
    _Float16* eW1  = (_Float16*)alloc(NLE * FW * 2);
    _Float16* eW2  = (_Float16*)alloc(NLE * FW * 2);
    _Float16* d1Wq = (_Float16*)alloc(NLD * PW * 2);
    _Float16* d1Wk = (_Float16*)alloc(NLD * PW * 2);
    _Float16* d1Wv = (_Float16*)alloc(NLD * PW * 2);
    _Float16* d1Wo = (_Float16*)alloc(NLD * PW * 2);
    _Float16* d2Wq = (_Float16*)alloc(NLD * PW * 2);
    _Float16* d2Wk = (_Float16*)alloc(NLD * PW * 2);
    _Float16* d2Wv = (_Float16*)alloc(NLD * PW * 2);
    _Float16* d2Wo = (_Float16*)alloc(NLD * PW * 2);
    _Float16* dW1  = (_Float16*)alloc(NLD * FW * 2);
    _Float16* dW2  = (_Float16*)alloc(NLD * FW * 2);

    // ---- weight conversion / repack (all stored N-major for transB path) ----
    {
        long long pTot = (long long)NLE * PW;
        int pg = (int)((pTot + 255) / 256);
        pack_proj_f16<<<pg, 256, 0, stream>>>(enc_Wq, eWq, NLE);
        pack_proj_f16<<<pg, 256, 0, stream>>>(enc_Wk, eWk, NLE);
        pack_proj_f16<<<pg, 256, 0, stream>>>(enc_Wv, eWv, NLE);
        pack_proj_f16<<<pg, 256, 0, stream>>>(dec1_Wq, d1Wq, NLD);
        pack_proj_f16<<<pg, 256, 0, stream>>>(dec1_Wk, d1Wk, NLD);
        pack_proj_f16<<<pg, 256, 0, stream>>>(dec1_Wv, d1Wv, NLD);
        pack_proj_f16<<<pg, 256, 0, stream>>>(dec2_Wq, d2Wq, NLD);
        pack_proj_f16<<<pg, 256, 0, stream>>>(dec2_Wk, d2Wk, NLD);
        pack_proj_f16<<<pg, 256, 0, stream>>>(dec2_Wv, d2Wv, NLD);
        cvt_t_f32_f16<<<pg, 256, 0, stream>>>(enc_Wo, eWo, DMODEL, DMODEL, NLE);
        cvt_t_f32_f16<<<pg, 256, 0, stream>>>(dec1_Wo, d1Wo, DMODEL, DMODEL, NLD);
        cvt_t_f32_f16<<<pg, 256, 0, stream>>>(dec2_Wo, d2Wo, DMODEL, DMODEL, NLD);
        long long fTot = (long long)NLE * FW;
        int fg = (int)((fTot + 255) / 256);
        cvt_t_f32_f16<<<fg, 256, 0, stream>>>(enc_W1, eW1, DMODEL, DFF, NLE);
        cvt_t_f32_f16<<<fg, 256, 0, stream>>>(enc_W2, eW2, DFF, DMODEL, NLE);
        cvt_t_f32_f16<<<fg, 256, 0, stream>>>(dec_W1, dW1, DMODEL, DFF, NLD);
        cvt_t_f32_f16<<<fg, 256, 0, stream>>>(dec_W2, dW2, DFF, DMODEL, NLD);
    }

    // ---- embeddings ----
    {
        int eg = (int)((AELEM + 255) / 256);
        embed_gather<<<eg, 256, 0, stream>>>(src_ids, src_emb, x32, x16, AELEM);
        embed_gather<<<eg, 256, 0, stream>>>(tgt_ids, tgt_emb, y32, y16, AELEM);
    }

    const long long TD  = (long long)TSEQ * DMODEL;
    const long long TT2 = (long long)TSEQ * TSEQ;
    const float isdk = 0.125f;                       // 1/sqrt(64)

    auto attention = [&](const _Float16* inq, const _Float16* inkv,
                         const _Float16* Wq, const _Float16* Wk,
                         const _Float16* Wv, const _Float16* Wo, int causal) {
        launch_gemm(stream, inq, Wq, (float*)0, q16, (const float*)0,
                    (int)NTOK, DMODEL, DMODEL, DMODEL, DMODEL, DMODEL,
                    0, 0, 0, 0, 0, 0, 1, 1, 1, 0, 1.0f);
        launch_gemm(stream, inkv, Wk, (float*)0, k16, (const float*)0,
                    (int)NTOK, DMODEL, DMODEL, DMODEL, DMODEL, DMODEL,
                    0, 0, 0, 0, 0, 0, 1, 1, 1, 0, 1.0f);
        launch_gemm(stream, inkv, Wv, (float*)0, v16, (const float*)0,
                    (int)NTOK, DMODEL, DMODEL, DMODEL, DMODEL, DMODEL,
                    0, 0, 0, 0, 0, 0, 1, 1, 1, 0, 1.0f);
        // scores[b,h,q,s] = (Qp . Kp^T) / sqrt(dk)   (B = Kp, transB)
        launch_gemm(stream, q16, k16, sc32, (_Float16*)0, (const float*)0,
                    TSEQ, TSEQ, DHEAD, DMODEL, DMODEL, TSEQ,
                    TD, DHEAD, TD, DHEAD,
                    (long long)NH * TT2, TT2,
                    NB * NH, NH, 1, 0, isdk);
        softmax_q<<<dim3(NB * NH, TSEQ / 256), 256, 0, stream>>>(
            sc32, at16, TSEQ, TSEQ, causal);
        // out[b,h,q,v] = attn . Vp   (non-trans B; lands concat-heads)
        launch_gemm(stream, at16, v16, (float*)0, ao16, (const float*)0,
                    TSEQ, DHEAD, TSEQ, TSEQ, DMODEL, DMODEL,
                    (long long)NH * TT2, TT2, TD, DHEAD, TD, DHEAD,
                    NB * NH, NH, 0, 0, 1.0f);
        launch_gemm(stream, ao16, Wo, t32, (_Float16*)0, (const float*)0,
                    (int)NTOK, DMODEL, DMODEL, DMODEL, DMODEL, DMODEL,
                    0, 0, 0, 0, 0, 0, 1, 1, 1, 0, 1.0f);
    };

    auto layer_norm = [&](const float* X, const float* R,
                          float* O32, _Float16* O16) {
        ln_stats<<<NB, 256, 0, stream>>>(X, R, stats, (int)TD);
        ln_apply<<<dim3(64, NB), 256, 0, stream>>>(X, R, stats, O32, O16, (int)TD);
    };

    // ---- encoder ----
    for (int l = 0; l < NLE; l++) {
        attention(x16, x16, eWq + (size_t)l * PW, eWk + (size_t)l * PW,
                  eWv + (size_t)l * PW, eWo + (size_t)l * PW, 0);
        layer_norm(x32, t32, x32, x16);
        launch_gemm(stream, x16, eW1 + (size_t)l * FW, (float*)0, h16,
                    enc_b1 + (size_t)l * DFF, (int)NTOK, DFF, DMODEL,
                    DMODEL, DMODEL, DFF, 0, 0, 0, 0, 0, 0, 1, 1, 1, 1, 1.0f);
        launch_gemm(stream, h16, eW2 + (size_t)l * FW, f32b, (_Float16*)0,
                    enc_b2 + (size_t)l * DMODEL, (int)NTOK, DMODEL, DFF,
                    DFF, DFF, DMODEL, 0, 0, 0, 0, 0, 0, 1, 1, 1, 0, 1.0f);
        layer_norm(x32, f32b, x32, x16);
    }

    // ---- decoder ----
    for (int l = 0; l < NLD; l++) {
        attention(y16, y16, d1Wq + (size_t)l * PW, d1Wk + (size_t)l * PW,
                  d1Wv + (size_t)l * PW, d1Wo + (size_t)l * PW, 1);
        layer_norm(y32, t32, y32, y16);
        attention(y16, x16, d2Wq + (size_t)l * PW, d2Wk + (size_t)l * PW,
                  d2Wv + (size_t)l * PW, d2Wo + (size_t)l * PW, 0);
        layer_norm(y32, t32, y32, y16);
        launch_gemm(stream, y16, dW1 + (size_t)l * FW, (float*)0, h16,
                    dec_b1 + (size_t)l * DFF, (int)NTOK, DFF, DMODEL,
                    DMODEL, DMODEL, DFF, 0, 0, 0, 0, 0, 0, 1, 1, 1, 1, 1.0f);
        launch_gemm(stream, h16, dW2 + (size_t)l * FW, f32b, (_Float16*)0,
                    dec_b2 + (size_t)l * DMODEL, (int)NTOK, DMODEL, DFF,
                    DFF, DFF, DMODEL, 0, 0, 0, 0, 0, 0, 1, 1, 1, 0, 1.0f);
        layer_norm(y32, f32b, y32, y16);
    }

    (void)hipMemcpyAsync(d_out, y32, (size_t)AELEM * 4,
                         hipMemcpyDeviceToDevice, stream);
}
